// Encoder_59219009077683
// MI455X (gfx1250) — matrix operand order repaired and checked by reading, not compiled, not run on hardware
//
#include <hip/hip_runtime.h>
#include <hip/hip_bf16.h>
#include <math.h>

// ---------------------------------------------------------------------------
// Swin block on MI455X (gfx1250): all GEMMs via v_wmma_f32_16x16x32_bf16,
// A-tile staging via CDNA5 async global->LDS copies, double-buffered LDS.
// ---------------------------------------------------------------------------

typedef __bf16 bf16;
typedef __bf16 v16bf __attribute__((ext_vector_type(16)));
typedef __bf16 v8bf  __attribute__((ext_vector_type(8)));
typedef float  v8f   __attribute__((ext_vector_type(8)));

#define TOK    100352      // B*H*W = 32*56*56
#define CDIM   384
#define NHEAD  12
#define NWH    24576       // 2048 windows * 12 heads
#define ATTN_SCALE 0.17677669529663687f   // 32^-0.5

// --- CDNA5 async global->LDS (ASYNCcnt) -----------------------------------
__device__ __forceinline__ void async_g2l_b128(unsigned lds_byte_addr,
                                               const void* sbase,
                                               unsigned byte_off) {
  // GVS mode: mem = SGPR64 + VGPR_i32; VDST supplies the LDS byte address.
  asm volatile("global_load_async_to_lds_b128 %0, %1, %2 offset:0"
               :
               : "v"(lds_byte_addr), "v"(byte_off), "s"(sbase)
               : "memory");
}

__device__ __forceinline__ void wait_async0() {
#if __has_builtin(__builtin_amdgcn_s_wait_asynccnt)
  __builtin_amdgcn_s_wait_asynccnt(0);
#else
  asm volatile("s_wait_asynccnt 0x0" ::: "memory");
#endif
}

// ---------------------------------------------------------------------------
__global__ __launch_bounds__(256) void cvt_f32_bf16(const float* __restrict__ in,
                                                    bf16* __restrict__ out, long n) {
  long i = (long)blockIdx.x * blockDim.x + threadIdx.x;
  const long stride = (long)gridDim.x * blockDim.x;
  for (; i < n; i += stride) out[i] = (bf16)in[i];
}

// ---------------------------------------------------------------------------
// Tiled BF16 GEMM, C = A(MxK) * B(KxN) + bias, f32 accumulate.
// Block tile 128x128, K-step 32, double-buffered LDS. 8 waves; each wave owns
// a 64x32 subtile = 4x2 WMMA 16x16 tiles.
// A tile: async global->LDS (2 x B128 per thread).  B tile: registers +
// transposed store lB[n][k] so B-fragments are K-contiguous.
// EPI: 0 = QKV (gathered A rows from shifted/windowed x, scatter-store to
// q/k/v head layout), 1/3 = plain f32 store, 2 = exact GELU -> bf16 store.
// ---------------------------------------------------------------------------
template<int EPI>
__global__ __launch_bounds__(256) void gemm_wmma_bf16(
    const bf16* __restrict__ A, const bf16* __restrict__ B,
    const float* __restrict__ bias,
    float* __restrict__ outF, bf16* __restrict__ outH,
    int M, int N, int K)
{
  __shared__ bf16 lA[2][128][40];   // padded stride: 40*2B = 80B
  __shared__ bf16 lB[2][128][40];   // stored transposed: lB[n][k]

  const int tid  = threadIdx.x;
  const int lane = tid & 31;
  const int wave = tid >> 5;
  const int wm   = wave >> 2;    // 0..1 : wave M offset (wm*64)
  const int wn   = wave & 3;     // 0..3 : wave N offset (wn*32)
  const int l16  = lane & 15;
  const int lh   = lane >> 4;    // lane half

  const int m0 = blockIdx.y * 128;
  const int n0 = blockIdx.x * 128;

  // --- A staging thread mapping: row = tid/2, 16 bf16 per thread ---
  const int ar = tid >> 1;
  const int ac = (tid & 1) * 16;
  long arow;
  if (EPI == 0) {
    // fused cyclic shift (-3,-3) + window partition gather:
    const int gm  = m0 + ar;
    const int b_  = gm / 49, tok = gm % 49;
    const int bb  = b_ >> 6, w_ = b_ & 63;
    const int wy  = w_ >> 3, wx = w_ & 7;
    int hs = wy * 7 + tok / 7 + 3; if (hs >= 56) hs -= 56;
    int ws = wx * 7 + tok % 7 + 3; if (ws >= 56) ws -= 56;
    arow = ((long)bb * 56 + hs) * 56 + ws;
  } else {
    arow = m0 + ar;
  }
  const unsigned abase_off = (unsigned)((arow * (long)K + ac) * 2);  // bytes
  const unsigned ldsA0 = (unsigned)(size_t)&lA[0][ar][ac];
  const unsigned ldsA1 = (unsigned)(size_t)&lA[1][ar][ac];

  // --- B staging thread mapping: k row = tid/8, 16 cols per thread ---
  const int bk = tid >> 3;
  const int bn = (tid & 7) * 16;

  v8f acc[4][2];
#pragma unroll
  for (int i = 0; i < 4; ++i)
#pragma unroll
    for (int j = 0; j < 2; ++j)
#pragma unroll
      for (int e = 0; e < 8; ++e) acc[i][j][e] = 0.0f;

  // ---- stage helpers (buf selects double-buffer half) ----
  auto stage = [&](int buf, int k0) {
    const unsigned la = buf ? ldsA1 : ldsA0;
    async_g2l_b128(la,      A, abase_off + (unsigned)(k0 * 2));
    async_g2l_b128(la + 16, A, abase_off + (unsigned)(k0 * 2) + 16);
    const bf16* bptr = B + (long)(k0 + bk) * N + n0 + bn;
    v8bf t0 = *(const v8bf*)bptr;
    v8bf t1 = *(const v8bf*)(bptr + 8);
#pragma unroll
    for (int j = 0; j < 8; ++j) {
      lB[buf][bn + j][bk]     = t0[j];
      lB[buf][bn + 8 + j][bk] = t1[j];
    }
  };

  stage(0, 0);
  wait_async0();
  __syncthreads();

  int cur = 0;
  for (int k0 = 0; k0 < K; k0 += 32) {
    if (k0 + 32 < K) stage(cur ^ 1, k0 + 32);   // prefetch next tile

    // A frag (16-bit A 16x32 layout): lane half lh -> K = lh*8..+7 / 16+lh*8..+7
    v16bf afrag[4];
#pragma unroll
    for (int mt = 0; mt < 4; ++mt) {
      const bf16* ap = &lA[cur][wm * 64 + mt * 16 + l16][0];
      v8bf x0 = *(const v8bf*)(ap + lh * 8);
      v8bf x1 = *(const v8bf*)(ap + 16 + lh * 8);
#pragma unroll
      for (int e = 0; e < 8; ++e) { afrag[mt][e] = x0[e]; afrag[mt][e + 8] = x1[e]; }
    }
    // B frag (16-bit B 32x16): lane = col N, K = lh*8..+7 / 16+lh*8..+7
    v16bf bfrag[2];
#pragma unroll
    for (int nt = 0; nt < 2; ++nt) {
      const bf16* bp = &lB[cur][wn * 32 + nt * 16 + l16][0];
      v8bf x0 = *(const v8bf*)(bp + lh * 8);
      v8bf x1 = *(const v8bf*)(bp + 16 + lh * 8);
#pragma unroll
      for (int e = 0; e < 8; ++e) { bfrag[nt][e] = x0[e]; bfrag[nt][e + 8] = x1[e]; }
    }
#pragma unroll
    for (int mt = 0; mt < 4; ++mt)
#pragma unroll
      for (int nt = 0; nt < 2; ++nt)
        acc[mt][nt] = __builtin_amdgcn_wmma_f32_16x16x32_bf16(
            false, afrag[mt], false, bfrag[nt], (short)0, acc[mt][nt], false, false);

    wait_async0();          // next tile's async A copies landed in LDS
    __syncthreads();        // (compiler adds dscnt wait for B stores)
    cur ^= 1;
  }

  // --- epilogue: C/D layout: lane N = l16, VGPR i -> M = i + 8*lh ---
#pragma unroll
  for (int mt = 0; mt < 4; ++mt) {
#pragma unroll
    for (int nt = 0; nt < 2; ++nt) {
      const int n  = n0 + wn * 32 + nt * 16 + l16;
      const float bv = bias[n];
#pragma unroll
      for (int i = 0; i < 8; ++i) {
        const int m = m0 + wm * 64 + mt * 16 + lh * 8 + i;
        float v = acc[mt][nt][i] + bv;
        if (EPI == 0) {
          // scatter into qkv[s][b_][h][tok][d] (bf16)
          const int s = n / 384, rem = n % 384;
          const int h = rem >> 5, d = rem & 31;
          const int b_ = m / 49, tok = m % 49;
          const long dst = ((long)s * TOK + ((long)b_ * NHEAD + h) * 49 + tok) * 32 + d;
          outH[dst] = (bf16)v;
        } else if (EPI == 2) {
          const float g = 0.5f * v * (1.0f + erff(v * 0.70710678118654752f));
          outH[(long)m * N + n] = (bf16)g;
        } else {
          outF[(long)m * N + n] = v;
        }
      }
    }
  }
}

// ---------------------------------------------------------------------------
// Per (window, head) attention: scores = scale * q @ k^T  (WMMA bf16, 49x49
// padded to 64x64), + rel-pos bias + shift mask, softmax; faithful to the
// reference einsum, o[n,d] = v[n,d] * sum_k softmax(n,k).
// ---------------------------------------------------------------------------
__global__ __launch_bounds__(256) void attn_win(
    const bf16* __restrict__ qkv,    // [3][TOK/49][NH][49][32] flat
    const float* __restrict__ rpb,   // [169][12]
    bf16* __restrict__ o)            // [B_][NH][49][32] flat
{
  __shared__ bf16  qs[64][40];
  __shared__ bf16  ks[64][40];
  __shared__ float sc[64][65];
  __shared__ float rs[64];

  const int tid = threadIdx.x;
  const int bh  = blockIdx.x;
  const int b_  = bh / NHEAD;
  const int h   = bh % NHEAD;
  const long base = (long)bh * 49 * 32;     // (b_*NH + h) * 49 * 32
  const long seg  = (long)TOK * CDIM;
  const bf16* qg = qkv + base;
  const bf16* kg = qkv + seg + base;
  const bf16* vg = qkv + 2 * seg + base;

  {  // stage q,k (rows 49..63 zero-padded)
    const int row = tid >> 2;
    const int col = (tid & 3) * 8;
    v8bf zero;
#pragma unroll
    for (int e = 0; e < 8; ++e) zero[e] = (bf16)0.0f;
    if (row < 49) {
      *(v8bf*)&qs[row][col] = *(const v8bf*)(qg + row * 32 + col);
      *(v8bf*)&ks[row][col] = *(const v8bf*)(kg + row * 32 + col);
    } else {
      *(v8bf*)&qs[row][col] = zero;
      *(v8bf*)&ks[row][col] = zero;
    }
  }
  __syncthreads();

  const int lane = tid & 31;
  const int wave = tid >> 5;
  const int l16  = lane & 15;
  const int lh   = lane >> 4;
#pragma unroll
  for (int tt = 0; tt < 2; ++tt) {    // 16 tiles / 8 waves
    const int t  = wave * 2 + tt;
    const int mt = t >> 2, nt = t & 3;
    v16bf af, bfv;
    const bf16* ap = &qs[mt * 16 + l16][0];
    v8bf a0 = *(const v8bf*)(ap + lh * 8);
    v8bf a1 = *(const v8bf*)(ap + 16 + lh * 8);
    const bf16* bp = &ks[nt * 16 + l16][0];    // B[k=d][n=j] == k[j][d]
    v8bf b0 = *(const v8bf*)(bp + lh * 8);
    v8bf b1 = *(const v8bf*)(bp + 16 + lh * 8);
#pragma unroll
    for (int e = 0; e < 8; ++e) {
      af[e] = a0[e];  af[e + 8]  = a1[e];
      bfv[e] = b0[e]; bfv[e + 8] = b1[e];
    }
    v8f acc;
#pragma unroll
    for (int e = 0; e < 8; ++e) acc[e] = 0.0f;
    acc = __builtin_amdgcn_wmma_f32_16x16x32_bf16(false, af, false, bfv,
                                                  (short)0, acc, false, false);
#pragma unroll
    for (int i = 0; i < 8; ++i)
      sc[mt * 16 + lh * 8 + i][nt * 16 + l16] = acc[i] * ATTN_SCALE;
  }
  __syncthreads();

  if (tid < 49) {   // per-row softmax with on-the-fly bias + shift mask
    const int q  = tid;
    const int r1 = q / 7, c1 = q % 7;
    const int w_ = b_ & 63;
    const int wy = w_ >> 3, wx = w_ & 7;
    const int hq = wy * 7 + r1, wq = wx * 7 + c1;
    const int rq = (hq < 49 ? 0 : (hq < 53 ? 1 : 2)) * 3
                 + (wq < 49 ? 0 : (wq < 53 ? 1 : 2));
    float mx = -1e30f;
    for (int j = 0; j < 49; ++j) {
      const int r2 = j / 7, c2 = j % 7;
      const int idx = (r1 - r2 + 6) * 13 + (c1 - c2 + 6);
      float s = sc[q][j] + rpb[idx * NHEAD + h];
      const int hk = wy * 7 + r2, wk = wx * 7 + c2;
      const int rk = (hk < 49 ? 0 : (hk < 53 ? 1 : 2)) * 3
                   + (wk < 49 ? 0 : (wk < 53 ? 1 : 2));
      if (rk != rq) s = -1e30f;
      sc[q][j] = s;
      mx = fmaxf(mx, s);
    }
    float se = 0.0f;
    for (int j = 0; j < 49; ++j) { const float e = __expf(sc[q][j] - mx); sc[q][j] = e; se += e; }
    const float inv = 1.0f / se;
    float ssum = 0.0f;
    for (int j = 0; j < 49; ++j) ssum += sc[q][j] * inv;
    rs[q] = ssum;
  }
  __syncthreads();

  for (int idx = tid; idx < 49 * 32; idx += 256)
    o[base + idx] = (bf16)((float)vg[idx] * rs[idx >> 5]);
}

// ---------------------------------------------------------------------------
// window reverse + roll(+3,+3) + LayerNorm(norm1) + residual.  One block per
// output token; 128 threads x 3 channels.
// ---------------------------------------------------------------------------
__global__ __launch_bounds__(128) void postattn_ln(
    const float* __restrict__ y,     // [B_*49][384] projection output
    const float* __restrict__ xin,   // original x (skip)
    const float* __restrict__ g, const float* __restrict__ bta,
    float* __restrict__ x2f, bf16* __restrict__ x2h)
{
  __shared__ float red[128];
  const int t   = blockIdx.x;
  const int tid = threadIdx.x;
  const int b   = t / 3136;
  const int sp  = t % 3136;
  const int hh  = sp / 56, ww = sp % 56;
  const int hsf = (hh + 53) % 56;           // un-roll: xw[(hh-3) mod 56]
  const int wsf = (ww + 53) % 56;
  const long yrow = ((long)b * 64 + (hsf / 7) * 8 + (wsf / 7)) * 49
                  + (hsf % 7) * 7 + (wsf % 7);
  const float* yp = y + yrow * CDIM;
  const float v0 = yp[tid], v1 = yp[tid + 128], v2 = yp[tid + 256];
  red[tid] = v0 + v1 + v2;
  __syncthreads();
  for (int off = 64; off > 0; off >>= 1) { if (tid < off) red[tid] += red[tid + off]; __syncthreads(); }
  const float mean = red[0] * (1.0f / 384.0f);
  __syncthreads();
  const float d0 = v0 - mean, d1 = v1 - mean, d2 = v2 - mean;
  red[tid] = d0 * d0 + d1 * d1 + d2 * d2;
  __syncthreads();
  for (int off = 64; off > 0; off >>= 1) { if (tid < off) red[tid] += red[tid + off]; __syncthreads(); }
  const float rstd = rsqrtf(red[0] * (1.0f / 384.0f) + 1e-5f);
  const long o0 = (long)t * CDIM;
#pragma unroll
  for (int e = 0; e < 3; ++e) {
    const int c = tid + e * 128;
    const float dv = (e == 0 ? d0 : (e == 1 ? d1 : d2));
    const float r  = dv * rstd * g[c] + bta[c] + xin[o0 + c];
    x2f[o0 + c] = r;
    x2h[o0 + c] = (bf16)r;
  }
}

// ---------------------------------------------------------------------------
// out = skip2 + LayerNorm(norm2)(h)
// ---------------------------------------------------------------------------
__global__ __launch_bounds__(128) void final_ln(
    const float* __restrict__ hbuf, const float* __restrict__ skip2,
    const float* __restrict__ g, const float* __restrict__ bta,
    float* __restrict__ out)
{
  __shared__ float red[128];
  const int t   = blockIdx.x;
  const int tid = threadIdx.x;
  const float* hp = hbuf + (long)t * CDIM;
  const float v0 = hp[tid], v1 = hp[tid + 128], v2 = hp[tid + 256];
  red[tid] = v0 + v1 + v2;
  __syncthreads();
  for (int off = 64; off > 0; off >>= 1) { if (tid < off) red[tid] += red[tid + off]; __syncthreads(); }
  const float mean = red[0] * (1.0f / 384.0f);
  __syncthreads();
  const float d0 = v0 - mean, d1 = v1 - mean, d2 = v2 - mean;
  red[tid] = d0 * d0 + d1 * d1 + d2 * d2;
  __syncthreads();
  for (int off = 64; off > 0; off >>= 1) { if (tid < off) red[tid] += red[tid + off]; __syncthreads(); }
  const float rstd = rsqrtf(red[0] * (1.0f / 384.0f) + 1e-5f);
  const long o0 = (long)t * CDIM;
#pragma unroll
  for (int e = 0; e < 3; ++e) {
    const int c = tid + e * 128;
    const float dv = (e == 0 ? d0 : (e == 1 ? d1 : d2));
    out[o0 + c] = skip2[o0 + c] + dv * rstd * g[c] + bta[c];
  }
}

// ---------------------------------------------------------------------------
extern "C" void kernel_launch(void* const* d_in, const int* in_sizes, int n_in,
                              void* d_out, int out_size, void* d_ws, size_t ws_size,
                              hipStream_t stream) {
  const float* x      = (const float*)d_in[0];
  const float* qkv_w  = (const float*)d_in[1];
  const float* qkv_b  = (const float*)d_in[2];
  const float* out_w  = (const float*)d_in[3];
  const float* out_b  = (const float*)d_in[4];
  const float* rpb    = (const float*)d_in[5];
  const float* n1g    = (const float*)d_in[6];
  const float* n1b    = (const float*)d_in[7];
  const float* n2g    = (const float*)d_in[8];
  const float* n2b    = (const float*)d_in[9];
  const float* mlp_w1 = (const float*)d_in[10];
  const float* mlp_b1 = (const float*)d_in[11];
  const float* mlp_w2 = (const float*)d_in[12];
  const float* mlp_b2 = (const float*)d_in[13];
  float* out = (float*)d_out;

  char* p = (char*)d_ws;
  auto carve = [&](size_t bytes) { char* r = p; p += (bytes + 255) & ~(size_t)255; return r; };
  bf16*  xh    = (bf16*)carve((size_t)TOK * CDIM * 2);
  bf16*  wqkvh = (bf16*)carve((size_t)CDIM * 1152 * 2);
  bf16*  wouth = (bf16*)carve((size_t)CDIM * CDIM * 2);
  bf16*  w1h   = (bf16*)carve((size_t)CDIM * 1536 * 2);
  bf16*  w2h   = (bf16*)carve((size_t)1536 * CDIM * 2);
  bf16*  qkvh  = (bf16*)carve((size_t)3 * TOK * CDIM * 2);
  bf16*  oh    = (bf16*)carve((size_t)TOK * CDIM * 2);   // attn out; reused as x2 bf16
  float* y     = (float*)carve((size_t)TOK * CDIM * 4);  // proj out; reused as MLP2 out
  float* x2f   = (float*)carve((size_t)TOK * CDIM * 4);
  bf16*  h1h   = (bf16*)carve((size_t)TOK * 1536 * 2);
  bf16*  x2h = oh;     // alias: oh dead after out-proj GEMM
  float* h2  = y;      // alias: y dead after postattn_ln

  // bf16 staging
  cvt_f32_bf16<<<2048, 256, 0, stream>>>(x,      xh,    (long)TOK * CDIM);
  cvt_f32_bf16<<<512,  256, 0, stream>>>(qkv_w,  wqkvh, 384L * 1152);
  cvt_f32_bf16<<<256,  256, 0, stream>>>(out_w,  wouth, 384L * 384);
  cvt_f32_bf16<<<512,  256, 0, stream>>>(mlp_w1, w1h,   384L * 1536);
  cvt_f32_bf16<<<512,  256, 0, stream>>>(mlp_w2, w2h,   1536L * 384);

  // QKV (fused shift + window gather), scatter to head layout
  gemm_wmma_bf16<0><<<dim3(9, 784), 256, 0, stream>>>(
      xh, wqkvh, qkv_b, nullptr, qkvh, TOK, 1152, 384);
  // attention per (window, head)
  attn_win<<<NWH, 256, 0, stream>>>(qkvh, rpb, oh);
  // out projection (flat o == reference's head-untransposed reshape)
  gemm_wmma_bf16<1><<<dim3(3, 784), 256, 0, stream>>>(
      oh, wouth, out_b, y, nullptr, TOK, 384, 384);
  // window reverse + roll + LN1 + skip
  postattn_ln<<<TOK, 128, 0, stream>>>(y, x, n1g, n1b, x2f, x2h);
  // MLP
  gemm_wmma_bf16<2><<<dim3(12, 784), 256, 0, stream>>>(
      x2h, w1h, mlp_b1, nullptr, h1h, TOK, 1536, 384);
  gemm_wmma_bf16<3><<<dim3(3, 784), 256, 0, stream>>>(
      h1h, w2h, mlp_b2, h2, nullptr, TOK, 384, 1536);
  // LN2 + skip2
  final_ln<<<TOK, 128, 0, stream>>>(h2, x2f, n2g, n2b, out);
}
